// GRU_43654047597257
// MI455X (gfx1250) — matrix operand order repaired and checked by reading, not compiled, not run on hardware
//
#include <hip/hip_runtime.h>
#include <hip/hip_bf16.h>

typedef __attribute__((ext_vector_type(16))) __bf16 v16bf;
typedef __attribute__((ext_vector_type(8)))  __bf16 v8bf;
typedef __attribute__((ext_vector_type(8)))  float  v8f;

#define NB      32
#define SEQ     512
#define M_ROWS  (NB*SEQ)      // 16384
#define EMB     300
#define EMBP    320           // padded K for input GEMM (10 x 32)
#define HID     200
#define HIDP    224           // padded K for recurrent GEMM (7 x 32)
#define G3      600           // 3*HID
#define G3P     608           // padded N (38 x 16)
#define GI_COLS 1200          // fwd 0..599, bwd 600..1199
#define OUTC    400
#define LABELS  10
#define NTHR    608           // 19 waves

// ---------- WMMA helpers (gfx1250 bf16, f32 accumulate) ----------
__device__ __forceinline__ v8f wmma_bf16(v16bf a, v16bf b, v8f c) {
  return __builtin_amdgcn_wmma_f32_16x16x32_bf16(false, a, false, b, (short)0, c,
                                                 false, false);
}

// A-matrix 16x32 bf16 per-lane fragment: row = lane&15,
// k = k0 + (lane>>4)*16 .. +15 (contiguous, matches B)
__device__ __forceinline__ v16bf ldA(const __bf16* p /* lane's row+chunk base */) {
  v8bf lo = *(const v8bf*)p;
  v8bf hi = *(const v8bf*)(p + 8);
  return __builtin_shufflevector(lo, hi, 0,1,2,3,4,5,6,7,8,9,10,11,12,13,14,15);
}

// ---------- K1a: embedding gather -> bf16 padded [M_ROWS][EMBP] ----------
__global__ void k_embed(const int* __restrict__ text, const float* __restrict__ table,
                        __bf16* __restrict__ embp) {
  int idx = blockIdx.x * blockDim.x + threadIdx.x;
  int m = idx / EMBP, k = idx % EMBP;
  float v = 0.f;
  if (k < EMB) { int tok = text[m]; v = table[(size_t)tok * EMB + k]; }
  embp[idx] = (__bf16)v;
}

// ---------- K1b: concat+pad input weights -> bf16 [GI_COLS][EMBP] ----------
__global__ void k_wcat(const float* __restrict__ wf, const float* __restrict__ wb,
                       __bf16* __restrict__ wcat) {
  int idx = blockIdx.x * blockDim.x + threadIdx.x;
  int g = idx / EMBP, k = idx % EMBP;
  float v = 0.f;
  if (k < EMB) v = (g < G3) ? wf[g * EMB + k] : wb[(g - G3) * EMB + k];
  wcat[idx] = (__bf16)v;
}

// ---------- K2: input projection GEMM  gi[16384,1200] = emb @ wcat^T + b ----------
// one 16x16 output tile per wave; K = 320 (10 WMMA steps)
__global__ void k_gemm_in(const __bf16* __restrict__ embp, const __bf16* __restrict__ wcat,
                          const float* __restrict__ bihf, const float* __restrict__ bihb,
                          float* __restrict__ gi) {
  int wave = (blockIdx.x * blockDim.x + threadIdx.x) >> 5;   // 76800 waves
  int lane = threadIdx.x & 31;
  int tM = wave / 75, tN = wave % 75;
  int row = lane & 15, half = lane >> 4;

  const __bf16* arow = embp + (size_t)(tM * 16 + row) * EMBP + half * 16;
  const __bf16* brow = wcat + (size_t)(tN * 16 + row) * EMBP + half * 16;

  v8f c = {};
#pragma unroll
  for (int k0 = 0; k0 < EMBP; k0 += 32) {
    v16bf a = ldA(arow + k0);
    v16bf b = *(const v16bf*)(brow + k0);   // B 32x16: 32 contiguous bytes/lane
    c = wmma_bf16(a, b, c);
  }
  int g = tN * 16 + row;
  float bias = (g < G3) ? bihf[g] : bihb[g - G3];
  float* out = gi + (size_t)(tM * 16 + half * 8) * GI_COLS + g;
#pragma unroll
  for (int r = 0; r < 8; ++r) out[(size_t)r * GI_COLS] = c[r] + bias;
}

// ---------- K3: serial GRU scan, one block per direction ----------
// LDS: whh bf16 [G3P][HIDP] (272KB) + h bf16 [NB][HIDP] + gh bf16 [NB][G3P]
__global__ void __launch_bounds__(NTHR)
k_scan(const float* __restrict__ whhF, const float* __restrict__ whhB,
       const float* __restrict__ bhhF, const float* __restrict__ bhhB,
       const float* __restrict__ h0F,  const float* __restrict__ h0B,
       const float* __restrict__ gi,   float* __restrict__ outc) {
  extern __shared__ char smem[];
  __bf16* whh = (__bf16*)smem;                                    // [G3P][HIDP]
  __bf16* hsh = (__bf16*)(smem + G3P * HIDP * 2);                 // [NB][HIDP]
  __bf16* ghb = (__bf16*)(smem + G3P * HIDP * 2 + NB * HIDP * 2); // [NB][G3P]

  const int dir = blockIdx.x;
  const float* whhg = dir ? whhB : whhF;
  const float* bhh  = dir ? bhhB : bhhF;
  const float* h0   = dir ? h0B  : h0F;
  const int tid = threadIdx.x;

  // preload padded recurrent weights into LDS (once; L2->LDS)
  for (int i = tid; i < G3P * HIDP; i += NTHR) {
    int g = i / HIDP, k = i % HIDP;
    float v = (g < G3 && k < HID) ? whhg[g * HID + k] : 0.f;
    whh[i] = (__bf16)v;
  }
  // init hidden state (bf16 matmul copy + exact f32 in registers)
  for (int i = tid; i < NB * HIDP; i += NTHR) {
    int m = i / HIDP, k = i % HIDP;
    float v = (k < HID) ? (h0[m * HID + k] - 0.5f) * (1.0f / HID) : 0.f;
    hsh[i] = (__bf16)v;
  }
  float hreg[11];
#pragma unroll
  for (int u = 0; u < 11; ++u) {
    int idx = tid + u * NTHR;
    if (idx < NB * HID) {
      int m = idx / HID, j = idx % HID;
      hreg[u] = (h0[m * HID + j] - 0.5f) * (1.0f / HID);
    }
  }
  __syncthreads();

  const int lane = tid & 31, wv = tid >> 5;
  const int row = lane & 15, half = lane >> 4;
  const int n0 = wv * 2, n1 = wv * 2 + 1;          // 19 waves cover 38 N-tiles
  const __bf16* a0p = hsh + row * HIDP + half * 16;          // M-tile 0 (rows 0..15)
  const __bf16* a1p = hsh + (16 + row) * HIDP + half * 16;   // M-tile 1 (rows 16..31)
  const __bf16* b0p = whh + (n0 * 16 + row) * HIDP + half * 16;
  const __bf16* b1p = whh + (n1 * 16 + row) * HIDP + half * 16;
  const float* gibase = gi + dir * G3;

  for (int t = 0; t < SEQ; ++t) {
    // ---- gh = h @ whh^T (32x600), all operands from LDS ----
    v8f c00 = {}, c01 = {}, c10 = {}, c11 = {};
#pragma unroll
    for (int k0 = 0; k0 < HIDP; k0 += 32) {
      v16bf a0 = ldA(a0p + k0);
      v16bf a1 = ldA(a1p + k0);
      v16bf b0 = *(const v16bf*)(b0p + k0);
      v16bf b1 = *(const v16bf*)(b1p + k0);
      c00 = wmma_bf16(a0, b0, c00);
      c01 = wmma_bf16(a0, b1, c01);
      c10 = wmma_bf16(a1, b0, c10);
      c11 = wmma_bf16(a1, b1, c11);
    }
#pragma unroll
    for (int r = 0; r < 8; ++r) {
      int m0 = r + half * 8;
      ghb[m0 * G3P + n0 * 16 + row]        = (__bf16)c00[r];
      ghb[m0 * G3P + n1 * 16 + row]        = (__bf16)c01[r];
      ghb[(16 + m0) * G3P + n0 * 16 + row] = (__bf16)c10[r];
      ghb[(16 + m0) * G3P + n1 * 16 + row] = (__bf16)c11[r];
    }
    __syncthreads();

    // ---- gates: r,z sigmoid; n tanh; h update (f32 state in registers) ----
    const int s = dir ? (SEQ - 1 - t) : t;
#pragma unroll
    for (int u = 0; u < 11; ++u) {
      int idx = tid + u * NTHR;
      if (idx < NB * HID) {
        int m = idx / HID, j = idx % HID;
        const float* girow = gibase + (size_t)(m * SEQ + s) * GI_COLS;
        float ir = girow[j], iz = girow[HID + j], in = girow[2 * HID + j];
        float hr = (float)ghb[m * G3P + j]           + bhh[j];
        float hz = (float)ghb[m * G3P + HID + j]     + bhh[HID + j];
        float hn = (float)ghb[m * G3P + 2 * HID + j] + bhh[2 * HID + j];
        float rg = 1.f / (1.f + __expf(-(ir + hr)));
        float zg = 1.f / (1.f + __expf(-(iz + hz)));
        float ng = tanhf(in + rg * hn);
        float hnew = (1.f - zg) * ng + zg * hreg[u];
        hreg[u] = hnew;
        hsh[m * HIDP + j] = (__bf16)hnew;
        outc[(size_t)(m * SEQ + s) * OUTC + dir * HID + j] = hnew;
      }
    }
    __syncthreads();
  }
}

// ---------- K4: final linear [16384,400] x [400,10] ----------
__global__ void k_linear(const float* __restrict__ outc, const float* __restrict__ lw,
                         const float* __restrict__ lb, float* __restrict__ preds) {
  int idx = blockIdx.x * blockDim.x + threadIdx.x;
  if (idx >= M_ROWS * LABELS) return;
  int m = idx / LABELS, l = idx % LABELS;
  const float* x = outc + (size_t)m * OUTC;
  const float* w = lw + l * OUTC;
  float acc = lb[l];
#pragma unroll 4
  for (int c = 0; c < OUTC; ++c) acc += x[c] * w[c];
  preds[idx] = acc;
}

extern "C" void kernel_launch(void* const* d_in, const int* in_sizes, int n_in,
                              void* d_out, int out_size, void* d_ws, size_t ws_size,
                              hipStream_t stream) {
  (void)in_sizes; (void)n_in; (void)out_size; (void)ws_size;
  const int*   text  = (const int*)d_in[0];
  const float* table = (const float*)d_in[1];
  const float* wihf  = (const float*)d_in[2];
  const float* whhf  = (const float*)d_in[3];
  const float* bihf  = (const float*)d_in[4];
  const float* bhhf  = (const float*)d_in[5];
  const float* wihb  = (const float*)d_in[6];
  const float* whhb  = (const float*)d_in[7];
  const float* bihb  = (const float*)d_in[8];
  const float* bhhb  = (const float*)d_in[9];
  const float* linw  = (const float*)d_in[10];
  const float* linb  = (const float*)d_in[11];
  const float* h0f   = (const float*)d_in[12];
  const float* h0b   = (const float*)d_in[13];
  float* preds = (float*)d_out;

  char* ws = (char*)d_ws;
  size_t off = 0;
  auto alloc = [&](size_t bytes) {
    char* p = ws + off;
    off += (bytes + 255) & ~(size_t)255;
    return p;
  };
  __bf16* embp = (__bf16*)alloc((size_t)M_ROWS * EMBP * 2);   // 10.5 MB
  __bf16* wcat = (__bf16*)alloc((size_t)GI_COLS * EMBP * 2);  // 0.77 MB
  float*  gi   = (float*) alloc((size_t)M_ROWS * GI_COLS * 4);// 78.6 MB
  float*  outc = (float*) alloc((size_t)M_ROWS * OUTC * 4);   // 26.2 MB

  k_embed<<<(M_ROWS * EMBP) / 256, 256, 0, stream>>>(text, table, embp);
  k_wcat<<<(GI_COLS * EMBP) / 256, 256, 0, stream>>>(wihf, wihb, wcat);
  k_gemm_in<<<9600, 256, 0, stream>>>(embp, wcat, bihf, bihb, gi);

  const int smemBytes = G3P * HIDP * 2 + NB * HIDP * 2 + NB * G3P * 2; // 325,632 B
  hipFuncSetAttribute((const void*)k_scan,
                      hipFuncAttributeMaxDynamicSharedMemorySize, smemBytes);
  k_scan<<<2, NTHR, smemBytes, stream>>>(whhf, whhb, bhhf, bhhb, h0f, h0b, gi, outc);

  k_linear<<<(M_ROWS * LABELS + 255) / 256, 256, 0, stream>>>(outc, linw, linb, preds);
}
